// SparseTransformer_27745488732195
// MI455X (gfx1250) — compile-verified, matrix operand order checked by reading, not run on hardware
//
#include <hip/hip_runtime.h>
#include <cstddef>
#include <cstdint>

// ---------------------------------------------------------------------------
// Problem constants (from reference): S=8192, B=4, D=512, H=8, WIN=64, L=6
// ---------------------------------------------------------------------------
#define SEQ   8192
#define BATCH 4
#define DM    512
#define NH    8
#define HDIM  64
#define WINSZ 64
#define NLAY  6
#define NWIN  (SEQ / WINSZ)      // 128
#define ROWS  (SEQ * BATCH)      // 32768 rows of the (S*B, D) activation matrix
#define D3    (3 * DM)           // 1536
#define DFF   (4 * DM)           // 2048

typedef __attribute__((ext_vector_type(16))) __bf16 v16bf;
typedef __attribute__((ext_vector_type(8)))  float  v8f;

union FragAB { v16bf v; uint4 q[2]; };

__device__ inline unsigned short f2bf(float f) {
  union { float f; unsigned int u; } c; c.f = f;
  unsigned int u = c.u;
  unsigned int r = u + 0x7FFFu + ((u >> 16) & 1u);     // round-nearest-even
  if ((u & 0x7F800000u) == 0x7F800000u) r = u;         // Inf/NaN passthrough
  return (unsigned short)(r >> 16);
}

__device__ inline v8f v8f_zero() {
  v8f z;
#pragma unroll
  for (int i = 0; i < 8; ++i) z[i] = 0.f;
  return z;
}

// Low 32 bits of a generic (flat) LDS pointer == byte offset within the
// wave's LDS allocation (ISA 10.2: in the LDS aperture the low 32 address bits give the LDS address).
__device__ inline unsigned lds_byte_off(const void* p) {
  return (unsigned)(unsigned long long)p;
}

// Async global->LDS 16B-per-lane copy (GLOBAL_LOAD_ASYNC_TO_LDS_B128,
// tracked by ASYNCcnt). Inline asm: portable across ROCm/upstream toolchains.
__device__ inline void async_ld16(unsigned ldsOff, unsigned long long gaddr) {
  asm volatile("global_load_async_to_lds_b128 %0, %1, off"
               :: "v"(ldsOff), "v"(gaddr)
               : "memory");
}

__device__ inline void wait_async8() {
#if __has_builtin(__builtin_amdgcn_s_wait_asynccnt)
  __builtin_amdgcn_s_wait_asynccnt(8);
#else
  asm volatile("s_wait_asynccnt 0x8" ::: "memory");
#endif
}
__device__ inline void wait_async0() {
#if __has_builtin(__builtin_amdgcn_s_wait_asynccnt)
  __builtin_amdgcn_s_wait_asynccnt(0);
#else
  asm volatile("s_wait_asynccnt 0x0" ::: "memory");
#endif
}

// Load one 16x32 bf16 WMMA A/B fragment from a row-major bf16 matrix.
// Lane L<16 holds row (row0+L), K = {k0..k0+7, k0+16..k0+23};
// lane L>=16 holds row (row0+L-16), K = {k0+8..k0+15, k0+24..k0+31}.
__device__ inline v16bf load_frag(const unsigned short* __restrict__ base,
                                  int stride, int row0, int k0,
                                  int rsel, int koff) {
  FragAB f;
  const unsigned short* p = base + (size_t)(row0 + rsel) * stride + k0 + koff;
  f.q[0] = *(const uint4*)(p);
  f.q[1] = *(const uint4*)(p + 16);
  return f.v;
}

// ---------------------------------------------------------------------------
// WMMA GEMM:  C(MxN) = A(MxK) * W(NxK)^T + bias, optional ReLU, out f32/bf16.
// Block = 4 waves -> 128x128 tile, wave tile 64x64 (16 accumulators).
// K is consumed in 32-steps; each step's 128x32 A and B tiles are staged into
// double-buffered LDS with global_load_async_to_lds_b128 (ASYNCcnt pipeline):
//   - 8 async copies per wave per step (16B/lane), in-order completion, so
//     s_wait_asynccnt<=8 after issuing the next buffer drains the current one.
//   - LDS row stride 40 bf16 (80B): lane stride 20 banks -> conflict-free
//     ds_load_b128 fragment reads.
// ---------------------------------------------------------------------------
#define TSTRIDE   40                        // bf16 elements per LDS tile row
#define TILEBYTES (128 * TSTRIDE * 2)       // 10240 B per 128x32 tile

template <bool RELU, bool OUTBF>
__global__ __launch_bounds__(128)
void gemm_wmma(const unsigned short* __restrict__ A,
               const unsigned short* __restrict__ W,
               const float* __restrict__ bias,
               void* __restrict__ Cout,
               int M, int N, int K) {
  __shared__ __align__(16) unsigned short Asmem[2 * 128 * TSTRIDE];
  __shared__ __align__(16) unsigned short Bsmem[2 * 128 * TSTRIDE];

  const int lane = threadIdx.x & 31;
  const int wave = threadIdx.x >> 5;
  const int tr = blockIdx.y * 128;
  const int tc = blockIdx.x * 128;
  const int wr = (wave >> 1) * 64;   // this wave's rows within the block tile
  const int wc = (wave & 1) * 64;    // this wave's cols within the block tile
  const int rsel = lane & 15;
  const int koff = (lane >> 4) << 3;

  // Async-copy lane mapping: one instruction moves 8 rows x 4 chunks (16B).
  const int lr = lane >> 2;          // row within 8-row group
  const int lc = (lane & 3) * 8;     // bf16 chunk offset within the 32-wide K

  unsigned long long ga[4], gb[4];
  unsigned la[4], lb[4];
  const unsigned aBase = lds_byte_off(Asmem);
  const unsigned bBase = lds_byte_off(Bsmem);
#pragma unroll
  for (int i = 0; i < 4; ++i) {
    const int r = wave * 32 + i * 8 + lr;        // rows 0..127 across 4 waves
    ga[i] = (unsigned long long)(const void*)(A + (size_t)(tr + r) * K + lc);
    gb[i] = (unsigned long long)(const void*)(W + (size_t)(tc + r) * K + lc);
    la[i] = aBase + (unsigned)(r * TSTRIDE + lc) * 2u;
    lb[i] = bBase + (unsigned)(r * TSTRIDE + lc) * 2u;
  }

  const int nk = K >> 5;

  // Prologue: stage k-step 0 into buffer 0.
#pragma unroll
  for (int i = 0; i < 4; ++i) async_ld16(la[i], ga[i]);
#pragma unroll
  for (int i = 0; i < 4; ++i) async_ld16(lb[i], gb[i]);

  v8f acc[16];
#pragma unroll
  for (int i = 0; i < 16; ++i) acc[i] = v8f_zero();

  for (int t = 0; t < nk; ++t) {
    __syncthreads();  // all waves done *reading* the buffer we refill below

    if (t + 1 < nk) {
      const unsigned nxtOff = (unsigned)((t + 1) & 1) * (unsigned)TILEBYTES;
      const unsigned gk = (unsigned)(t + 1) * 64u;  // 32 bf16 = 64 bytes per step
#pragma unroll
      for (int i = 0; i < 4; ++i) async_ld16(la[i] + nxtOff, ga[i] + gk);
#pragma unroll
      for (int i = 0; i < 4; ++i) async_ld16(lb[i] + nxtOff, gb[i] + gk);
      wait_async8();  // oldest 8 (current buffer) complete, next 8 in flight
    } else {
      wait_async0();
    }
    __syncthreads();  // every wave's current-buffer copies are visible

    const unsigned short* Acur = Asmem + (size_t)(t & 1) * 128 * TSTRIDE;
    const unsigned short* Bcur = Bsmem + (size_t)(t & 1) * 128 * TSTRIDE;

    FragAB a[4], b[4];
#pragma unroll
    for (int i = 0; i < 4; ++i) {
      const unsigned short* p = Acur + (wr + i * 16 + rsel) * TSTRIDE + koff;
      a[i].q[0] = *(const uint4*)(p);
      a[i].q[1] = *(const uint4*)(p + 16);
    }
#pragma unroll
    for (int j = 0; j < 4; ++j) {
      const unsigned short* p = Bcur + (wc + j * 16 + rsel) * TSTRIDE + koff;
      b[j].q[0] = *(const uint4*)(p);
      b[j].q[1] = *(const uint4*)(p + 16);
    }
#pragma unroll
    for (int i = 0; i < 4; ++i)
#pragma unroll
      for (int j = 0; j < 4; ++j)
        acc[i * 4 + j] = __builtin_amdgcn_wmma_f32_16x16x32_bf16(
            false, a[i].v, false, b[j].v, (short)0, acc[i * 4 + j], false, false);
  }

#pragma unroll
  for (int i = 0; i < 4; ++i) {
#pragma unroll
    for (int j = 0; j < 4; ++j) {
      const int col = tc + wc + j * 16 + rsel;
      const float bv = bias[col];
#pragma unroll
      for (int e = 0; e < 8; ++e) {
        const int row = tr + wr + i * 16 + e + koff;
        float v = acc[i * 4 + j][e] + bv;
        if (RELU) v = v > 0.f ? v : 0.f;
        if (OUTBF)
          ((unsigned short*)Cout)[(size_t)row * N + col] = f2bf(v);
        else
          ((float*)Cout)[(size_t)row * N + col] = v;
      }
    }
  }
  (void)M;
}

// ---------------------------------------------------------------------------
// Windowed attention: one block (4 waves) per (window, batch, head).
// Q already scaled (folded into Wqkv/bqkv at cast time).
// Q/K staged row-major in LDS; V staged transposed so P*V B-fragments are
// contiguous. Softmax done entirely in the WMMA accumulator layout with
// wave32 xor-shuffles (masks 8/4/2/1 stay inside each 16-lane half).
// ---------------------------------------------------------------------------
__global__ __launch_bounds__(128)
void attn_win_kernel(const unsigned short* __restrict__ qkv,  // (ROWS, 1536) bf16
                     unsigned short* __restrict__ o) {        // (ROWS, 512)  bf16
  constexpr int LS = 72;  // padded bf16 row stride (144B, 16B-aligned)
  __shared__ __align__(16) unsigned short Qs[WINSZ * LS];
  __shared__ __align__(16) unsigned short Ks[WINSZ * LS];
  __shared__ __align__(16) unsigned short Vt[HDIM * LS];
  __shared__ __align__(16) unsigned short Ps[4 * 16 * LS];

  const int bx  = blockIdx.x;
  const int h   = bx & (NH - 1);
  const int b   = (bx >> 3) & (BATCH - 1);
  const int win = bx >> 5;
  const int tid  = threadIdx.x;
  const int lane = tid & 31;
  const int wave = tid >> 5;

  // Stage Q, K, V^T tiles (bf16) into LDS.
  for (int idx = tid; idx < WINSZ * HDIM; idx += 128) {
    const int j = idx >> 6;
    const int d = idx & 63;
    const size_t base =
        (size_t)((win * WINSZ + j) * BATCH + b) * D3 + h * HDIM + d;
    Qs[j * LS + d] = qkv[base];
    Ks[j * LS + d] = qkv[base + DM];
    Vt[d * LS + j] = qkv[base + 2 * DM];
  }
  __syncthreads();

  const int rsel = lane & 15;
  const int koff = (lane >> 4) << 3;
  const int q0 = wave * 16;  // this wave's 16 query rows

  // scores(16x64) = Q(16x64) * K^T   (K-dim = head dim 64, two 32-steps)
  v8f sc[4];
#pragma unroll
  for (int t = 0; t < 4; ++t) sc[t] = v8f_zero();
#pragma unroll
  for (int k0 = 0; k0 < HDIM; k0 += 32) {
    const v16bf qa = load_frag(Qs, LS, q0, k0, rsel, koff);
#pragma unroll
    for (int t = 0; t < 4; ++t) {
      const v16bf kb = load_frag(Ks, LS, t * 16, k0, rsel, koff);
      sc[t] = __builtin_amdgcn_wmma_f32_16x16x32_bf16(
          false, qa, false, kb, (short)0, sc[t], false, false);
    }
  }

  // Row softmax in fragment layout: row (e + koff) lives across lanes of one
  // 16-lane half (col = lane&15) and across the 4 column fragments.
#pragma unroll
  for (int e = 0; e < 8; ++e) {
    float m = fmaxf(fmaxf(sc[0][e], sc[1][e]), fmaxf(sc[2][e], sc[3][e]));
#pragma unroll
    for (int off = 8; off >= 1; off >>= 1) m = fmaxf(m, __shfl_xor(m, off, 32));
    float s = 0.f;
#pragma unroll
    for (int t = 0; t < 4; ++t) {
      const float ev = __expf(sc[t][e] - m);
      sc[t][e] = ev;
      s += ev;
    }
#pragma unroll
    for (int off = 8; off >= 1; off >>= 1) s += __shfl_xor(s, off, 32);
    const float inv = 1.f / s;
#pragma unroll
    for (int t = 0; t < 4; ++t) sc[t][e] *= inv;
  }

  // Spill P (bf16) to this wave's private LDS slab; re-fragment as WMMA A.
  unsigned short* P = Ps + wave * 16 * LS;
#pragma unroll
  for (int t = 0; t < 4; ++t)
#pragma unroll
    for (int e = 0; e < 8; ++e)
      P[(e + koff) * LS + t * 16 + rsel] = f2bf(sc[t][e]);

  // out(16x64) = P(16x64) * V(64x64); V columns are Vt rows (contiguous).
  v8f ov[4];
#pragma unroll
  for (int t = 0; t < 4; ++t) ov[t] = v8f_zero();
#pragma unroll
  for (int k0 = 0; k0 < WINSZ; k0 += 32) {
    const v16bf pa = load_frag(P, LS, 0, k0, rsel, koff);
#pragma unroll
    for (int t = 0; t < 4; ++t) {
      const v16bf vb = load_frag(Vt, LS, t * 16, k0, rsel, koff);
      ov[t] = __builtin_amdgcn_wmma_f32_16x16x32_bf16(
          false, pa, false, vb, (short)0, ov[t], false, false);
    }
  }

#pragma unroll
  for (int t = 0; t < 4; ++t)
#pragma unroll
    for (int e = 0; e < 8; ++e) {
      const int qrow = q0 + e + koff;
      const size_t row = (size_t)((win * WINSZ + qrow) * BATCH + b);
      o[row * DM + h * HDIM + t * 16 + rsel] = f2bf(ov[t][e]);
    }
}

// ---------------------------------------------------------------------------
// Fused residual + LayerNorm, one wave per 512-wide row (16 elems/lane on
// wave32). Writes f32 residual stream and bf16 copy for the next GEMM.
// ---------------------------------------------------------------------------
__global__ __launch_bounds__(256)
void add_ln_kernel(const float* __restrict__ x, const float* __restrict__ ybr,
                   const float* __restrict__ g, const float* __restrict__ be,
                   float* __restrict__ xo, unsigned short* __restrict__ xob) {
  const int lane = threadIdx.x & 31;
  const int wave = threadIdx.x >> 5;
  const size_t row = (size_t)blockIdx.x * 8 + wave;
  const float* px = x + row * DM;
  const float* py = ybr + row * DM;

  float v[16];
  float s = 0.f;
#pragma unroll
  for (int t = 0; t < 16; ++t) {
    v[t] = px[t * 32 + lane] + py[t * 32 + lane];
    s += v[t];
  }
#pragma unroll
  for (int off = 16; off >= 1; off >>= 1) s += __shfl_xor(s, off, 32);
  const float m = s * (1.f / DM);

  float s2 = 0.f;
#pragma unroll
  for (int t = 0; t < 16; ++t) {
    const float d = v[t] - m;
    s2 += d * d;
  }
#pragma unroll
  for (int off = 16; off >= 1; off >>= 1) s2 += __shfl_xor(s2, off, 32);
  const float rstd = rsqrtf(s2 * (1.f / DM) + 1e-5f);

#pragma unroll
  for (int t = 0; t < 16; ++t) {
    const int c = t * 32 + lane;
    const float ov = (v[t] - m) * rstd * g[c] + be[c];
    xo[row * DM + c] = ov;
    xob[row * DM + c] = f2bf(ov);
  }
}

// ---------------------------------------------------------------------------
// Prep kernels: f32 -> bf16 casts; fold the 1/sqrt(HD) attention scale into
// the Q-rows of Wqkv/bqkv so the attention kernel never rescales.
// ---------------------------------------------------------------------------
__global__ void init_x_kernel(const float* __restrict__ s, float* __restrict__ x,
                              unsigned short* __restrict__ xb, size_t n) {
  size_t i = (size_t)blockIdx.x * blockDim.x + threadIdx.x;
  const size_t st = (size_t)gridDim.x * blockDim.x;
  for (; i < n; i += st) {
    const float v = s[i];
    x[i] = v;
    xb[i] = f2bf(v);
  }
}

__global__ void cast_kernel(const float* __restrict__ in,
                            unsigned short* __restrict__ out, size_t n) {
  size_t i = (size_t)blockIdx.x * blockDim.x + threadIdx.x;
  const size_t st = (size_t)gridDim.x * blockDim.x;
  for (; i < n; i += st) out[i] = f2bf(in[i]);
}

__global__ void cast_qkvw_kernel(const float* __restrict__ in,
                                 unsigned short* __restrict__ out, float sc,
                                 size_t n) {
  size_t i = (size_t)blockIdx.x * blockDim.x + threadIdx.x;
  const size_t st = (size_t)gridDim.x * blockDim.x;
  for (; i < n; i += st) {
    const size_t r = (i >> 9) % D3;  // K = DM = 512 = 1<<9; r = output row
    out[i] = f2bf(in[i] * (r < DM ? sc : 1.f));
  }
}

__global__ void scale_qkvb_kernel(const float* __restrict__ in,
                                  float* __restrict__ out, float sc, size_t n) {
  size_t i = (size_t)blockIdx.x * blockDim.x + threadIdx.x;
  const size_t st = (size_t)gridDim.x * blockDim.x;
  for (; i < n; i += st) {
    const size_t r = i % D3;
    out[i] = in[i] * (r < DM ? sc : 1.f);
  }
}

// ---------------------------------------------------------------------------
// Driver
// ---------------------------------------------------------------------------
extern "C" void kernel_launch(void* const* d_in, const int* in_sizes, int n_in,
                              void* d_out, int out_size, void* d_ws,
                              size_t ws_size, hipStream_t stream) {
  (void)in_sizes; (void)n_in; (void)out_size; (void)ws_size;
  const float* src  = (const float*)d_in[0];
  const float* Wqkv = (const float*)d_in[1];
  const float* bqkv = (const float*)d_in[2];
  const float* Wo   = (const float*)d_in[3];
  const float* bo   = (const float*)d_in[4];
  const float* W1   = (const float*)d_in[5];
  const float* bf1  = (const float*)d_in[6];
  const float* W2   = (const float*)d_in[7];
  const float* bf2  = (const float*)d_in[8];
  const float* g1   = (const float*)d_in[9];
  const float* be1  = (const float*)d_in[10];
  const float* g2   = (const float*)d_in[11];
  const float* be2  = (const float*)d_in[12];

  char* ws = (char*)d_ws;
  size_t off = 0;
  auto carve = [&](size_t bytes) -> char* {
    char* p = ws + off;
    off += (bytes + 255) & ~(size_t)255;
    return p;
  };
  float*          xf    = (float*)carve((size_t)ROWS * DM * 4);
  unsigned short* xb    = (unsigned short*)carve((size_t)ROWS * DM * 2);
  unsigned short* qkvb  = (unsigned short*)carve((size_t)ROWS * D3 * 2);
  unsigned short* ob    = (unsigned short*)carve((size_t)ROWS * DM * 2);
  float*          yf    = (float*)carve((size_t)ROWS * DM * 4);
  unsigned short* hb    = (unsigned short*)carve((size_t)ROWS * DFF * 2);
  unsigned short* Wqkvb = (unsigned short*)carve((size_t)NLAY * D3 * DM * 2);
  unsigned short* Wob   = (unsigned short*)carve((size_t)NLAY * DM * DM * 2);
  unsigned short* W1b   = (unsigned short*)carve((size_t)NLAY * DFF * DM * 2);
  unsigned short* W2b   = (unsigned short*)carve((size_t)NLAY * DM * DFF * 2);
  float*          bqs   = (float*)carve((size_t)NLAY * D3 * 4);

  const float scale = 0.125f;  // 1/sqrt(HDIM)

  // --- one-time-per-call prep (cheap; keeps launch deterministic) ---
  init_x_kernel<<<8192, 256, 0, stream>>>(src, xf, xb, (size_t)ROWS * DM);
  cast_qkvw_kernel<<<4608, 256, 0, stream>>>(Wqkv, Wqkvb, scale,
                                             (size_t)NLAY * D3 * DM);
  scale_qkvb_kernel<<<36, 256, 0, stream>>>(bqkv, bqs, scale,
                                            (size_t)NLAY * D3);
  cast_kernel<<<3072, 256, 0, stream>>>(Wo, Wob, (size_t)NLAY * DM * DM);
  cast_kernel<<<6144, 256, 0, stream>>>(W1, W1b, (size_t)NLAY * DFF * DM);
  cast_kernel<<<6144, 256, 0, stream>>>(W2, W2b, (size_t)NLAY * DM * DFF);

  // --- layers ---
  for (int l = 0; l < NLAY; ++l) {
    gemm_wmma<false, true><<<dim3(D3 / 128, ROWS / 128), 128, 0, stream>>>(
        xb, Wqkvb + (size_t)l * D3 * DM, bqs + (size_t)l * D3, qkvb,
        ROWS, D3, DM);
    attn_win_kernel<<<NWIN * BATCH * NH, 128, 0, stream>>>(qkvb, ob);
    gemm_wmma<false, false><<<dim3(DM / 128, ROWS / 128), 128, 0, stream>>>(
        ob, Wob + (size_t)l * DM * DM, bo + (size_t)l * DM, yf, ROWS, DM, DM);
    add_ln_kernel<<<ROWS / 8, 256, 0, stream>>>(xf, yf, g1 + (size_t)l * DM,
                                                be1 + (size_t)l * DM, xf, xb);
    gemm_wmma<true, true><<<dim3(DFF / 128, ROWS / 128), 128, 0, stream>>>(
        xb, W1b + (size_t)l * DFF * DM, bf1 + (size_t)l * DFF, hb,
        ROWS, DFF, DM);
    gemm_wmma<false, false><<<dim3(DM / 128, ROWS / 128), 128, 0, stream>>>(
        hb, W2b + (size_t)l * DM * DFF, bf2 + (size_t)l * DM, yf,
        ROWS, DM, DFF);
    add_ln_kernel<<<ROWS / 8, 256, 0, stream>>>(xf, yf, g2 + (size_t)l * DM,
                                                be2 + (size_t)l * DM, xf, xb);
  }

  hipMemcpyAsync(d_out, xf, (size_t)ROWS * DM * sizeof(float),
                 hipMemcpyDeviceToDevice, stream);
}
